// NeuralNetworkOnAEV_12506944766610
// MI455X (gfx1250) — matrix operand order repaired and checked by reading, not compiled, not run on hardware
//
#include <hip/hip_runtime.h>

// ---------------------------------------------------------------------------
// ANI-style per-species MLP over AEVs, CDNA5 (gfx1250) wave32.
// v_wmma_f32_16x16x32_bf16 hot loop fed by ds_load_b128, plus
// global_load_async_to_lds_b128 weight staging double-buffered in LDS and
// overlapped with compute (ASYNCcnt-synchronized).
// ---------------------------------------------------------------------------

typedef __attribute__((ext_vector_type(16))) __bf16 v16bf;
typedef __attribute__((ext_vector_type(8)))  float  v8f;

#define N_SPECIES 4
#define BMOL 128      // molecules
#define ATOMS 512     // atoms per molecule
#define AEV 384
#define H1 160
#define H2 128
#define H3 96
#define CELU_ALPHA 0.1f

// Activation row strides in bf16 elements (dword stride multiple of 4 for
// b128 alignment; not a multiple of 256 to avoid systematic bank aliasing)
#define SX0 392   // 384 + 8  -> 196 dwords
#define SY1 168   // 160 + 8  ->  84 dwords
#define SY2 136   // 128 + 8  ->  68 dwords
#define SY3 104   //  96 + 8  ->  52 dwords

#define WPAD 4    // LDS weight row pad (dwords): keeps 16B align, breaks banks

// LDS region layout in dwords (dynamic LDS base == offset 0: no static LDS)
#define OFF_SX  0
#define OFF_SY  (BMOL * (SX0 / 2))                 // 25088
#define OFF_SW0 (OFF_SY + BMOL * (SY1 / 2))        // 35840
#define OFF_SW1 (OFF_SW0 + H1 * (AEV / 2 + WPAD))  // 67200
#define LDS_DW  (OFF_SW1 + H2 * (H1 / 2 + WPAD))   // 77952 dwords = 311808 B

#define S_WAIT_ASYNC(n) asm volatile("s_wait_asynccnt " #n ::: "memory")

static __device__ inline unsigned short f32_to_bf16(float f) {
  unsigned int u = __float_as_uint(f);
  return (unsigned short)((u + 0x7FFFu + ((u >> 16) & 1u)) >> 16);  // RNE
}
static __device__ inline unsigned int pack_bf16x2(float lo, float hi) {
  return (unsigned int)f32_to_bf16(lo) | ((unsigned int)f32_to_bf16(hi) << 16);
}
static __device__ inline float bf16_to_f32(unsigned short h) {
  return __uint_as_float(((unsigned int)h) << 16);
}

// One lane: async copy 16B global -> LDS (tracked by ASYNCcnt).
static __device__ inline void async_ld_b128(unsigned lds_byte_off, const void* gaddr) {
  asm volatile("global_load_async_to_lds_b128 %0, %1, off"
               :: "v"(lds_byte_off), "v"(gaddr) : "memory");
}

union BF16Frag { v16bf v; unsigned int u[8]; };

// A-fragment (16x32 bf16, MxK) from LDS. Documented 16-bit A layout:
//   lanes 0-15 (M=lane):    V0..V3 = K{0,1..6,7},  V4..V7 = K{16..23}
//   lanes16-31 (M=lane-16): V0..V3 = K{8,9..14,15},V4..V7 = K{24..31}
// -> per lane, two contiguous runs of 4 dwords.
static __device__ inline BF16Frag load_a_frag(const unsigned int* lds,
                                              int strideU, int mbase, int kbase) {
  BF16Frag f;
  const int lane = (int)(threadIdx.x & 31u);
  const int lh   = lane >> 4;
  const unsigned int* p = lds + (mbase + (lane & 15)) * strideU + (kbase >> 1) + lh * 4;
  uint4 lo = *(const uint4*)(p);
  uint4 hi = *(const uint4*)(p + 8);
  f.u[0] = lo.x; f.u[1] = lo.y; f.u[2] = lo.z; f.u[3] = lo.w;
  f.u[4] = hi.x; f.u[5] = hi.y; f.u[6] = hi.z; f.u[7] = hi.w;
  return f;
}

// B-fragment (32x16 bf16, KxN) from LDS-staged weights (row-major [N][K]):
//   lanes 0-15 (N=lane): K=0..7,16..23; lanes16-31: K=8..15,24..31 (two runs of 4 dwords).
static __device__ inline BF16Frag load_b_frag(const unsigned int* sW,
                                              int rowU, int nbase, int kbase) {
  BF16Frag f;
  const int lane = (int)(threadIdx.x & 31u);
  const int lh   = lane >> 4;
  const unsigned int* p = sW + (nbase + (lane & 15)) * rowU + (kbase >> 1) + lh * 4;
  uint4 lo = *(const uint4*)(p);
  uint4 hi = *(const uint4*)(p + 8);
  f.u[0] = lo.x; f.u[1] = lo.y; f.u[2] = lo.z; f.u[3] = lo.w;
  f.u[4] = hi.x; f.u[5] = hi.y; f.u[6] = hi.z; f.u[7] = hi.w;
  return f;
}

// Issue async copies of one species' bf16 weight matrix [OUT][IN/2 dwords]
// into padded LDS region at dword offset ldsBase ([OUT][IN/2+WPAD] dwords).
// Does NOT wait; caller synchronizes with s_wait_asynccnt.
template <int IN, int OUT>
static __device__ void stage_weights_async(const unsigned int* __restrict__ Wg,
                                           int ldsBase) {
  constexpr int Q    = IN / 8;          // uint4 per row
  constexpr int NQ   = OUT * Q;         // total uint4 (multiple of 256 here)
  constexpr int ROWB = (IN / 2 + WPAD) * 4;
  for (int idx = (int)threadIdx.x; idx < NQ; idx += 256) {
    const int row = idx / Q;
    const int c4  = idx - row * Q;
    async_ld_b128((unsigned)(ldsBase * 4 + row * ROWB + c4 * 16),
                  (const void*)(Wg + (size_t)row * (IN / 2) + 4 * c4));
  }
}

// One dense layer: Y = act(X @ W^T + bias), bf16 in LDS, f32 accumulate.
// 8 waves; wave w owns M-tile w (M = 128 rows = 8 tiles of 16).
template <int IN, int OUT, int SIN, int SOUT, bool ACT>
static __device__ void mlp_layer(const unsigned int* __restrict__ Xlds,
                                 unsigned int* __restrict__ Ylds,
                                 const unsigned int* __restrict__ sW,
                                 const float* __restrict__ Bias) {
  const int wave  = (int)(threadIdx.x >> 5);
  const int lane  = (int)(threadIdx.x & 31u);
  const int lh    = lane >> 4;
  const int mbase = wave * 16;
  constexpr int NT   = OUT / 16;
  constexpr int KT   = IN / 32;
  constexpr int ROWU = IN / 2 + WPAD;

  v8f zero = {};
  v8f acc[NT];
#pragma unroll
  for (int n = 0; n < NT; ++n) acc[n] = zero;

  for (int kb = 0; kb < KT; ++kb) {
    BF16Frag a = load_a_frag(Xlds, SIN / 2, mbase, kb * 32);
#pragma unroll
    for (int n = 0; n < NT; ++n) {
      BF16Frag b = load_b_frag(sW, ROWU, n * 16, kb * 32);
      acc[n] = __builtin_amdgcn_wmma_f32_16x16x32_bf16(
          false, a.v, false, b.v, (short)0, acc[n], false, false);
    }
  }

  // Epilogue: C/D layout -> VGPR r: lanes0-15 M=r, lanes16-31 M=r+8, N=lane&15
  unsigned short* Yh = (unsigned short*)Ylds;
#pragma unroll
  for (int n = 0; n < NT; ++n) {
    const int N = n * 16 + (lane & 15);
    const float bv = Bias[N];
#pragma unroll
    for (int r = 0; r < 8; ++r) {
      const int M = mbase + r + (lh ? 8 : 0);
      float y = acc[n][r] + bv;
      if (ACT) y = (y > 0.f) ? y : CELU_ALPHA * (__expf(y * (1.0f / CELU_ALPHA)) - 1.f);
      Yh[M * SOUT + N] = f32_to_bf16(y);
    }
  }
}

__global__ __launch_bounds__(256) void ani_mlp_kernel(
    const float* __restrict__ fullaev, const int* __restrict__ species,
    const unsigned int* __restrict__ W0b, const float* __restrict__ b0,
    const unsigned int* __restrict__ W1b, const float* __restrict__ b1,
    const unsigned int* __restrict__ W2b, const float* __restrict__ b2,
    const unsigned int* __restrict__ W3b, const float* __restrict__ b3,
    float* __restrict__ part) {
  extern __shared__ unsigned int smem[];
  unsigned int* sX  = smem + OFF_SX;
  unsigned int* sY  = smem + OFF_SY;
  unsigned int* sW0 = smem + OFF_SW0;   // layer-0 W, later reused for layer-2 W
  unsigned int* sW1 = smem + OFF_SW1;   // layer-1 W

  const int a   = (int)blockIdx.x;
  const int s   = species[a];
  const int tid = (int)threadIdx.x;

  // Kick off async weight staging (W0: 30 ops/wave, W1: 10 ops/wave).
  stage_weights_async<AEV, H1>(W0b + (size_t)s * H1 * (AEV / 2), OFF_SW0);
  stage_weights_async<H1, H2>(W1b + (size_t)s * H2 * (H1 / 2), OFF_SW1);

  // Meanwhile: stage this atom's AEV slice [128 x 384] f32 -> bf16 LDS.
  constexpr int Q4 = AEV / 4;  // 96 float4 per row
  for (int idx = tid; idx < BMOL * Q4; idx += 256) {
    const int b = idx / Q4;
    const int p = idx - b * Q4;
    const float4 xv = *(const float4*)(fullaev + ((size_t)b * ATOMS + a) * AEV + 4 * p);
    unsigned int* dst = sX + b * (SX0 / 2) + 2 * p;
    dst[0] = pack_bf16x2(xv.x, xv.y);
    dst[1] = pack_bf16x2(xv.z, xv.w);
  }

  S_WAIT_ASYNC(10);          // in-order completion: W0 landed (W1 may fly)
  __syncthreads();

  // Layer 0: [128,384] -> [128,160]
  mlp_layer<AEV, H1, SX0, SY1, true>(sX, sY, sW0, b0 + s * H1);
  __syncthreads();           // sW0 no longer read by any wave

  // Stage layer-2 weights into the freed sW0 region (6 ops/wave).
  stage_weights_async<H2, H3>(W2b + (size_t)s * H3 * (H2 / 2), OFF_SW0);

  S_WAIT_ASYNC(6);           // W1 landed (W2 may fly)
  __syncthreads();

  // Layer 1: [128,160] -> [128,128]  (output reuses sX region)
  mlp_layer<H1, H2, SY1, SY2, true>(sY, sX, sW1, b1 + s * H2);

  S_WAIT_ASYNC(0);           // W2 landed (this wave)
  __syncthreads();           // ... and every other wave's portion

  // Layer 2: [128,128] -> [128,96]
  mlp_layer<H2, H3, SY2, SY3, true>(sX, sY, sW0, b2 + s * H3);
  __syncthreads();

  // Layer 3: [128,96] . W3[96] + b3 -> per-atom partial energy (packed bf16)
  if (tid < BMOL) {
    const unsigned int* w3 = W3b + s * (H3 / 2);
    float accv = b3[s];
#pragma unroll 8
    for (int j2 = 0; j2 < H3 / 2; ++j2) {
      const unsigned int xv = sY[tid * (SY3 / 2) + j2];
      const unsigned int wv = w3[j2];
      accv = fmaf(bf16_to_f32((unsigned short)(xv & 0xFFFFu)),
                  bf16_to_f32((unsigned short)(wv & 0xFFFFu)), accv);
      accv = fmaf(bf16_to_f32((unsigned short)(xv >> 16)),
                  bf16_to_f32((unsigned short)(wv >> 16)), accv);
    }
    part[(size_t)a * BMOL + tid] = accv;
  }
}

// Pack f32 weights into bf16 pairs (one-time per launch, deterministic).
__global__ __launch_bounds__(256) void pack_weights_kernel(
    const float* __restrict__ src, unsigned int* __restrict__ dst, int npairs) {
  const int i = (int)(blockIdx.x * blockDim.x + threadIdx.x);
  if (i < npairs) dst[i] = pack_bf16x2(src[2 * i], src[2 * i + 1]);
}

// Deterministic sum over atoms: out[b] = sum_a part[a][b]
__global__ __launch_bounds__(128) void reduce_atoms_kernel(
    const float* __restrict__ part, float* __restrict__ out) {
  const int b = (int)threadIdx.x;
  float s = 0.f;
  for (int a = 0; a < ATOMS; ++a) s += part[(size_t)a * BMOL + b];
  out[b] = s;
}

extern "C" void kernel_launch(void* const* d_in, const int* in_sizes, int n_in,
                              void* d_out, int out_size, void* d_ws, size_t ws_size,
                              hipStream_t stream) {
  (void)in_sizes; (void)n_in; (void)out_size; (void)ws_size;
  const float* fullaev = (const float*)d_in[0];
  const int*   species = (const int*)d_in[1];
  const float* W0 = (const float*)d_in[2];
  const float* b0 = (const float*)d_in[3];
  const float* W1 = (const float*)d_in[4];
  const float* b1 = (const float*)d_in[5];
  const float* W2 = (const float*)d_in[6];
  const float* b2 = (const float*)d_in[7];
  const float* W3 = (const float*)d_in[8];
  const float* b3 = (const float*)d_in[9];
  float* out = (float*)d_out;

  // Workspace layout (16B aligned): partials f32[512][128], then bf16 weights.
  char* ws = (char*)d_ws;
  float*        part = (float*)ws;
  const size_t  PART_B = (size_t)ATOMS * BMOL * sizeof(float);        // 262144
  unsigned int* W0b = (unsigned int*)(ws + PART_B);
  const int     W0P = N_SPECIES * H1 * (AEV / 2);                     // 122880 pairs
  unsigned int* W1b = W0b + W0P;
  const int     W1P = N_SPECIES * H2 * (H1 / 2);                      // 40960
  unsigned int* W2b = W1b + W1P;
  const int     W2P = N_SPECIES * H3 * (H2 / 2);                      // 24576
  unsigned int* W3b = W2b + W2P;
  const int     W3P = N_SPECIES * (H3 / 2);                           // 192

  pack_weights_kernel<<<(W0P + 255) / 256, 256, 0, stream>>>(W0, W0b, W0P);
  pack_weights_kernel<<<(W1P + 255) / 256, 256, 0, stream>>>(W1, W1b, W1P);
  pack_weights_kernel<<<(W2P + 255) / 256, 256, 0, stream>>>(W2, W2b, W2P);
  pack_weights_kernel<<<(W3P + 255) / 256, 256, 0, stream>>>(W3, W3b, W3P);

  const size_t lds_bytes = (size_t)LDS_DW * sizeof(unsigned int);  // ~305 KB

  ani_mlp_kernel<<<ATOMS, 256, lds_bytes, stream>>>(
      fullaev, species, W0b, b0, W1b, b1, W2b, b2, W3b, b3, part);
  reduce_atoms_kernel<<<1, BMOL, 0, stream>>>(part, out);
}
